// FourierNetworkOperator1D_86895778333270
// MI455X (gfx1250) — matrix operand order repaired and checked by reading, not compiled, not run on hardware
//
#include <hip/hip_runtime.h>
#include <math.h>

#define B_ 32
#define W_ 64
#define L_ 8192
#define MODES_ 16
#define NCH 16
#define CHL (L_ / NCH) // 512

typedef __attribute__((ext_vector_type(8)))  float  v8f;
typedef __attribute__((ext_vector_type(16))) __bf16 v16bf;
typedef __attribute__((ext_vector_type(8)))  __bf16 v8bf;

static __device__ __forceinline__ v16bf join16(v8bf lo, v8bf hi) {
  v16bf r;
#pragma unroll
  for (int i = 0; i < 8; i++) { r[i] = lo[i]; r[i + 8] = hi[i]; }
  return r;
}

// ---------------- basis tables ----------------
// FcT[k][l] = cos(2*pi*k*l/L), FsT[k][l] = -sin(...)       (forward rDFT, 16 modes)
// Bas2[l][k]    = s_k * cos(2*pi*k*l/L)   (s_0 = 1/L, s_k = 2/L)
// Bas2[l][16+k] = -s_k * sin(2*pi*k*l/L)  (irfft with 16 modes; Im(X0) drops via sin0=0)
__global__ void k_init_basis(__bf16* __restrict__ FcT, __bf16* __restrict__ FsT,
                             __bf16* __restrict__ Bas2) {
  int l = blockIdx.x * blockDim.x + threadIdx.x;
  if (l >= L_) return;
  for (int k = 0; k < MODES_; k++) {
    float th = 6.283185307179586f * (float)k * (float)l / (float)L_;
    float c = cosf(th), s = sinf(th);
    FcT[k * L_ + l] = (__bf16)c;
    FsT[k * L_ + l] = (__bf16)(-s);
    float sc = (k == 0 ? 1.0f : 2.0f) / (float)L_;
    Bas2[l * 32 + k]      = (__bf16)(sc * c);
    Bas2[l * 32 + 16 + k] = (__bf16)(-sc * s);
  }
}

// pre-round conv weights to bf16 once (all 4 layers)
__global__ void k_cvt_w(const float* __restrict__ cw, __bf16* __restrict__ cwb) {
  int i = blockIdx.x * blockDim.x + threadIdx.x; // < 4*64*64
  if (i < 4 * W_ * W_) cwb[i] = (__bf16)cw[i];
}

// ---------------- lifting 1x1 conv (DA=2 -> W), bf16 activations ----------------
__global__ void k_lift(const float* __restrict__ x, const float* __restrict__ Pw,
                       const float* __restrict__ Pb, __bf16* __restrict__ h) {
  int idx = blockIdx.x * blockDim.x + threadIdx.x; // < B*W*L
  int l = idx & (L_ - 1);
  int o = (idx >> 13) & (W_ - 1);
  int b = idx >> 19;
  float v = Pw[o * 2 + 0] * x[(b * 2 + 0) * L_ + l]
          + Pw[o * 2 + 1] * x[(b * 2 + 1) * L_ + l] + Pb[o];
  h[idx] = (__bf16)v;
}

// ---------------- forward DFT (16 modes) partials via bf16 WMMA ----------------
// rows = flattened (b,c) in 128 tiles of 16; K = l reduced in 16 chunks of 512
__global__ void __launch_bounds__(32)
k_dft_partial(const __bf16* __restrict__ h, const __bf16* __restrict__ FcT,
              const __bf16* __restrict__ FsT, float* __restrict__ part) {
  const int tile = blockIdx.x;   // 0..127
  const int chunk = blockIdx.y;  // 0..NCH-1
  const int lane = threadIdx.x;
  const int g = lane >> 4, q = lane & 15;
  const __bf16* rowp = h + (size_t)(tile * 16 + q) * L_;
  v8f accR = {}; v8f accI = {};
  const int l0 = chunk * CHL;
#pragma unroll 4
  for (int l = l0; l < l0 + CHL; l += 32) {
    // streaming read: prefetch ~1KB ahead (global_prefetch_b8)
    __builtin_prefetch(rowp + l + 512, 0, 1);
    // A operand: lane holds K = 16g+0..15 contiguous -> one 32B load
    v16bf A = *reinterpret_cast<const v16bf*>(rowp + l + 16 * g);
    // B operand: lane (column q) holds K = 16g+0..15, contiguous in mode-major basis
    v16bf Bc = *reinterpret_cast<const v16bf*>(FcT + (size_t)q * L_ + l + 16 * g);
    v16bf Bs = *reinterpret_cast<const v16bf*>(FsT + (size_t)q * L_ + l + 16 * g);
    accR = __builtin_amdgcn_wmma_f32_16x16x32_bf16(false, A, false, Bc, (short)0, accR, false, false);
    accI = __builtin_amdgcn_wmma_f32_16x16x32_bf16(false, A, false, Bs, (short)0, accI, false, false);
  }
  float* pR = part + ((size_t)(chunk * 128 + tile) * 2 + 0) * 256 + lane * 8;
  float* pI = part + ((size_t)(chunk * 128 + tile) * 2 + 1) * 256 + lane * 8;
#pragma unroll
  for (int j = 0; j < 8; j++) { pR[j] = accR[j]; pI[j] = accI[j]; }
}

// ---------------- deterministic chunk reduction ----------------
__global__ void k_dft_reduce(const float* __restrict__ part, float* __restrict__ Hr,
                             float* __restrict__ Hi) {
  int idx = blockIdx.x * blockDim.x + threadIdx.x; // < 2048*16
  if (idx >= 2048 * MODES_) return;
  int n = idx & 15, r = idx >> 4;
  int tile = r >> 4, m = r & 15;
  int pos = ((m >> 3) * 16 + n) * 8 + (m & 7); // C/D layout: lane=(m>>3)*16+n, vgpr=m&7
  float sR = 0.f, sI = 0.f;
  for (int c = 0; c < NCH; c++) {
    sR += part[((size_t)(c * 128 + tile) * 2 + 0) * 256 + pos];
    sI += part[((size_t)(c * 128 + tile) * 2 + 1) * 256 + pos];
  }
  Hr[r * MODES_ + n] = sR;
  Hi[r * MODES_ + n] = sI;
}

// ---------------- complex per-mode channel mixing (tiny, f32) ----------------
__global__ void __launch_bounds__(64)
k_mode_mix(const float* __restrict__ Hr, const float* __restrict__ Hi,
           const float* __restrict__ kwr, const float* __restrict__ kwi,
           __bf16* __restrict__ Coef) {
  const int b = blockIdx.x, k = blockIdx.y, o = threadIdx.x;
  __shared__ float hr[W_], hi[W_];
  hr[o] = Hr[(b * W_ + o) * MODES_ + k];
  hi[o] = Hi[(b * W_ + o) * MODES_ + k];
  __syncthreads();
  const float* wr = kwr + ((size_t)k * W_ + o) * W_;
  const float* wi = kwi + ((size_t)k * W_ + o) * W_;
  float gr = 0.f, gi = 0.f;
  for (int i = 0; i < W_; i++) {
    float a = wr[i], c = wi[i];
    gr += a * hr[i] - c * hi[i];
    gi += a * hi[i] + c * hr[i];
  }
  // A-operand row for inverse transform: [Gr(16) | Gi(16)]
  Coef[(size_t)(b * W_ + o) * 32 + k]      = (__bf16)gr;
  Coef[(size_t)(b * W_ + o) * 32 + 16 + k] = (__bf16)gi;
}

// ---------------- fused inverse-DFT + channel conv + bias + GELU (in place) ----------------
__global__ void __launch_bounds__(128)
k_fused_out(__bf16* __restrict__ h, const __bf16* __restrict__ Coef,
            const __bf16* __restrict__ Bas2, const __bf16* __restrict__ convWb,
            const float* __restrict__ convB) {
  const int b = blockIdx.x;
  const int lt = blockIdx.y;
  const int wave = threadIdx.x >> 5, lane = threadIdx.x & 31;
  const int g = lane >> 4, q = lane & 15;
  const int l0 = lt * 64 + wave * 16;

  // inverse-DFT B operand: lane (column l0+q) holds K = 16g+0..15, contiguous row of Bas2
  v16bf Bi = *reinterpret_cast<const v16bf*>(Bas2 + (size_t)(l0 + q) * 32 + 16 * g);

  // conv B operands: K = input channel (two halves of 32); column = l0+q
  v16bf Bh0, Bh1;
#pragma unroll
  for (int v = 0; v < 8; v++) {
#pragma unroll
    for (int jj = 0; jj < 2; jj++) {
      int c0 = 16 * g + 2 * v + jj;
      Bh0[2 * v + jj] = h[((size_t)(b * W_ + c0)) * L_ + l0 + q];
      Bh1[2 * v + jj] = h[((size_t)(b * W_ + c0 + 32)) * L_ + l0 + q];
    }
  }

  v8f acc[4];
#pragma unroll
  for (int mt = 0; mt < 4; mt++) {
    v8f c = {};
    { // inverse DFT: K=32 coefficients
      const __bf16* rp = Coef + (size_t)(b * W_ + mt * 16 + q) * 32;
      v16bf A = *reinterpret_cast<const v16bf*>(rp + 16 * g);
      c = __builtin_amdgcn_wmma_f32_16x16x32_bf16(false, A, false, Bi, (short)0, c, false, false);
    }
    { // channel conv: K=64 in two WMMAs, weights pre-rounded to bf16
      const __bf16* rp = convWb + (size_t)(mt * 16 + q) * W_;
      v16bf A0 = *reinterpret_cast<const v16bf*>(rp + 16 * g);
      v16bf A1 = *reinterpret_cast<const v16bf*>(rp + 32 + 16 * g);
      c = __builtin_amdgcn_wmma_f32_16x16x32_bf16(false, A0, false, Bh0, (short)0, c, false, false);
      c = __builtin_amdgcn_wmma_f32_16x16x32_bf16(false, A1, false, Bh1, (short)0, c, false, false);
    }
    acc[mt] = c;
  }

  // epilogue: bias + exact GELU; all h reads for this wave's columns already done -> in-place safe
#pragma unroll
  for (int mt = 0; mt < 4; mt++) {
#pragma unroll
    for (int j = 0; j < 8; j++) {
      int o = mt * 16 + 8 * g + j;
      float v = acc[mt][j] + convB[o];
      float gl = 0.5f * v * (1.0f + erff(v * 0.70710678118654752f));
      h[((size_t)(b * W_ + o)) * L_ + l0 + q] = (__bf16)gl;
    }
  }
}

// ---------------- final projection (DU=1) ----------------
__global__ void k_final(const __bf16* __restrict__ h, const float* __restrict__ Qw,
                        const float* __restrict__ Qb, float* __restrict__ out) {
  int idx = blockIdx.x * blockDim.x + threadIdx.x; // < B*L
  int l = idx & (L_ - 1), b = idx >> 13;
  float s = Qb[0];
#pragma unroll 8
  for (int c = 0; c < W_; c++) s += Qw[c] * (float)h[((size_t)(b * W_ + c)) * L_ + l];
  out[idx] = s;
}

extern "C" void kernel_launch(void* const* d_in, const int* in_sizes, int n_in,
                              void* d_out, int out_size, void* d_ws, size_t ws_size,
                              hipStream_t stream) {
  const float* x   = (const float*)d_in[0];
  const float* Pw  = (const float*)d_in[1];
  const float* Pb  = (const float*)d_in[2];
  const float* kwr = (const float*)d_in[3];
  const float* kwi = (const float*)d_in[4];
  const float* cw  = (const float*)d_in[5];
  const float* cb  = (const float*)d_in[6];
  const float* Qw  = (const float*)d_in[7];
  const float* Qb  = (const float*)d_in[8];
  float* out = (float*)d_out;

  char* ws = (char*)d_ws;
  size_t off = 0;
  __bf16* h    = (__bf16*)(ws + off); off += (size_t)B_ * W_ * L_ * 2;        // 32 MB
  __bf16* FcT  = (__bf16*)(ws + off); off += (size_t)MODES_ * L_ * 2;
  __bf16* FsT  = (__bf16*)(ws + off); off += (size_t)MODES_ * L_ * 2;
  __bf16* Bas2 = (__bf16*)(ws + off); off += (size_t)L_ * 32 * 2;
  float*  part = (float*)(ws + off);  off += (size_t)NCH * 128 * 2 * 256 * 4; // 4 MB
  float*  Hr   = (float*)(ws + off);  off += (size_t)2048 * MODES_ * 4;
  float*  Hi   = (float*)(ws + off);  off += (size_t)2048 * MODES_ * 4;
  __bf16* Coef = (__bf16*)(ws + off); off += (size_t)2048 * 32 * 2;
  __bf16* cwb  = (__bf16*)(ws + off); off += (size_t)4 * W_ * W_ * 2;
  (void)ws_size; (void)in_sizes; (void)n_in; (void)out_size;

  k_init_basis<<<(L_ + 127) / 128, 128, 0, stream>>>(FcT, FsT, Bas2);
  k_cvt_w<<<(4 * W_ * W_) / 256, 256, 0, stream>>>(cw, cwb);
  k_lift<<<(B_ * W_ * L_) / 256, 256, 0, stream>>>(x, Pw, Pb, h);

  for (int i = 0; i < 4; i++) {
    k_dft_partial<<<dim3(128, NCH), 32, 0, stream>>>(h, FcT, FsT, part);
    k_dft_reduce<<<(2048 * MODES_) / 256, 256, 0, stream>>>(part, Hr, Hi);
    k_mode_mix<<<dim3(B_, MODES_), W_, 0, stream>>>(
        Hr, Hi, kwr + (size_t)i * MODES_ * W_ * W_, kwi + (size_t)i * MODES_ * W_ * W_, Coef);
    k_fused_out<<<dim3(B_, L_ / 64), 128, 0, stream>>>(
        h, Coef, Bas2, cwb + (size_t)i * W_ * W_, cb + (size_t)i * W_);
  }

  k_final<<<(B_ * L_) / 256, 256, 0, stream>>>(h, Qw, Qb, out);
}
